// GPT2Model_45200235823413
// MI455X (gfx1250) — matrix operand order repaired and checked by reading, not compiled, not run on hardware
//
#include <hip/hip_runtime.h>
#include <hip/hip_bf16.h>
#include <math.h>

// ---- model constants (fixed by the reference) ----
#define TT   1024
#define DD   1024
#define NHH  16
#define HDD  64
#define LL   12
#define FF   4096
#define BB   2
#define MM   (BB*TT)      // 2048 token rows
#define VV   50257
#define SCALE 0.125f      // HD^-0.5

typedef __attribute__((ext_vector_type(16))) __bf16 v16bf;
typedef __attribute__((ext_vector_type(8)))  float  v8f;

__device__ __forceinline__ __bf16 f2bf(float f) {
  unsigned u = __builtin_bit_cast(unsigned, f);
  unsigned r = u + 0x7FFFu + ((u >> 16) & 1u);   // round-to-nearest-even
  return __builtin_bit_cast(__bf16, (unsigned short)(r >> 16));
}
__device__ __forceinline__ float bf2f(__bf16 h) {
  unsigned u = ((unsigned)__builtin_bit_cast(unsigned short, h)) << 16;
  return __builtin_bit_cast(float, u);
}
__device__ __forceinline__ __bf16 bflo(unsigned pk) {
  return __builtin_bit_cast(__bf16, (unsigned short)(pk & 0xffffu));
}
__device__ __forceinline__ __bf16 bfhi(unsigned pk) {
  return __builtin_bit_cast(__bf16, (unsigned short)(pk >> 16));
}

// CDNA5 async copy: global -> LDS, 16 bytes per lane, tracked by ASYNCcnt.
// VDST VGPR holds the LDS byte offset (relative to the wave's LDS base),
// which is the low 32 bits of the generic shared-memory pointer.
__device__ __forceinline__ void async_g2l_b128(unsigned lds_off, const void* gaddr) {
  asm volatile("global_load_async_to_lds_b128 %0, %1, off"
               :: "v"(lds_off), "v"(gaddr) : "memory");
}
__device__ __forceinline__ void wait_async0() {
  asm volatile("s_wait_asynccnt 0" ::: "memory");
}
// async loads complete in order: <=4 outstanding means the older batch landed
__device__ __forceinline__ void wait_async4() {
  asm volatile("s_wait_asynccnt 4" ::: "memory");
}
__device__ __forceinline__ unsigned lds_off32(const void* p) {
  return (unsigned)(size_t)p;
}

// =====================================================================
// fp32 -> bf16 bulk convert (weights, once per matrix per layer)
// =====================================================================
__global__ __launch_bounds__(256)
void cvt_f32_bf16_kernel(const float* __restrict__ src, __bf16* __restrict__ dst, int n) {
  int i = (blockIdx.x * 256 + threadIdx.x) * 8;
  if (i >= n) return;
  float4 a = *(const float4*)(src + i);
  float4 b = *(const float4*)(src + i + 4);
  unsigned short h[8];
  h[0] = __builtin_bit_cast(unsigned short, f2bf(a.x));
  h[1] = __builtin_bit_cast(unsigned short, f2bf(a.y));
  h[2] = __builtin_bit_cast(unsigned short, f2bf(a.z));
  h[3] = __builtin_bit_cast(unsigned short, f2bf(a.w));
  h[4] = __builtin_bit_cast(unsigned short, f2bf(b.x));
  h[5] = __builtin_bit_cast(unsigned short, f2bf(b.y));
  h[6] = __builtin_bit_cast(unsigned short, f2bf(b.z));
  h[7] = __builtin_bit_cast(unsigned short, f2bf(b.w));
  uint4 pk;
  pk.x = (unsigned)h[0] | ((unsigned)h[1] << 16);
  pk.y = (unsigned)h[2] | ((unsigned)h[3] << 16);
  pk.z = (unsigned)h[4] | ((unsigned)h[5] << 16);
  pk.w = (unsigned)h[6] | ((unsigned)h[7] << 16);
  *(uint4*)(dst + i) = pk;
}

// =====================================================================
// Token + position embedding: x[row, :] = wte[tok[row], :] + wpe[t, :]
// =====================================================================
__global__ __launch_bounds__(256)
void embed_kernel(const int* __restrict__ tok, const float* __restrict__ wte,
                  const float* __restrict__ wpe, float* __restrict__ x) {
  int row = blockIdx.x;
  int t = row % TT;
  int tk = tok[row];
  for (int c = threadIdx.x; c < DD; c += 256)
    x[(size_t)row * DD + c] = wte[(size_t)tk * DD + c] + wpe[(size_t)t * DD + c];
}

// =====================================================================
// LayerNorm over D=1024, fp32 in -> bf16 out (GEMM A operand)
// row mapping: x_row = x + (rowBase + blockIdx.x*rowStride)*D ; out compact
// =====================================================================
__global__ __launch_bounds__(256)
void layernorm_kernel(const float* __restrict__ x, const float* __restrict__ g,
                      const float* __restrict__ bta, __bf16* __restrict__ out,
                      int rowBase, int rowStride) {
  __shared__ float red[256];
  int row = blockIdx.x;
  int tid = threadIdx.x;
  const float* xr = x + ((size_t)rowBase + (size_t)row * rowStride) * DD;

  float v[4]; float s = 0.f;
#pragma unroll
  for (int i = 0; i < 4; i++) { v[i] = xr[tid + i * 256]; s += v[i]; }
  red[tid] = s; __syncthreads();
  for (int o = 128; o > 0; o >>= 1) { if (tid < o) red[tid] += red[tid + o]; __syncthreads(); }
  float mean = red[0] * (1.0f / DD);
  __syncthreads();
  float s2 = 0.f;
#pragma unroll
  for (int i = 0; i < 4; i++) { float d = v[i] - mean; s2 += d * d; }
  red[tid] = s2; __syncthreads();
  for (int o = 128; o > 0; o >>= 1) { if (tid < o) red[tid] += red[tid + o]; __syncthreads(); }
  float inv = rsqrtf(red[0] * (1.0f / DD) + 1e-5f);
#pragma unroll
  for (int i = 0; i < 4; i++) {
    int c = tid + i * 256;
    out[(size_t)row * DD + c] = f2bf((v[i] - mean) * inv * g[c] + bta[c]);
  }
}

// =====================================================================
// GEMM: C[M,N] = A_bf16[M,K] @ W_bf16[K,N] + bias[N], bf16 WMMA, f32 accum
// Block tile 128x128 (8 waves, each 32x64 = 2x4 of 16x16). K-step 32.
// Double-buffered CDNA5 async global->LDS pipeline:
//   issue tile i+1 -> s_wait_asynccnt 4 -> barrier -> 8x WMMA on tile i -> barrier
// EPI: 0 = bias -> bf16 out
//      1 = bias + exact GELU -> bf16 out
//      2 = bias + residual add into fp32 Xres (N == D)
//      3 = bias -> scatter into q/k/v [B,NH,T,HD] bf16 (reference d = hd*NH+nh)
// =====================================================================
template <int EPI>
__global__ __launch_bounds__(256)
void gemm_bf16_kernel(const __bf16* __restrict__ A, const __bf16* __restrict__ W,
                      const float* __restrict__ bias, float* __restrict__ Xres,
                      __bf16* __restrict__ OutB,
                      __bf16* __restrict__ qb, __bf16* __restrict__ kbuf,
                      __bf16* __restrict__ vbuf,
                      int M, int N, int K) {
  __shared__ __bf16 As[2][128][40];   // 128x32 tiles, +8 pad (row stride 80B, 16B aligned)
  __shared__ __bf16 Bs[2][32][136];   // 32x128 tiles, +8 pad (row stride 272B, 16B aligned)

  const int tid  = threadIdx.x;
  const int lane = tid & 31, wid = tid >> 5;
  const int wm = wid & 3, wn = wid >> 2;      // waves: 4 along M, 2 along N
  const int blockM = blockIdx.y * 128, blockN = blockIdx.x * 128;
  const int half = lane >> 4, l16 = lane & 15;

  v8f acc[2][4];
#pragma unroll
  for (int i = 0; i < 2; i++)
#pragma unroll
    for (int j = 0; j < 4; j++)
#pragma unroll
      for (int e = 0; e < 8; e++) acc[i][j][e] = 0.f;

  const int ar = tid >> 1, ac = (tid & 1) * 16;     // A loader: 128 rows x 32
  const int br = tid >> 3, bc = (tid & 7) * 16;     // B loader: 32 rows x 128
  const unsigned aoff0 = lds_off32(&As[0][ar][ac]);
  const unsigned aoff1 = lds_off32(&As[1][ar][ac]);
  const unsigned boff0 = lds_off32(&Bs[0][br][bc]);
  const unsigned boff1 = lds_off32(&Bs[1][br][bc]);
  const __bf16* gaBase = A + (size_t)(blockM + ar) * K + ac;
  const __bf16* gbCol  = W + blockN + bc;

  auto issue = [&](int buf, int kt) {
    const __bf16* ga = gaBase + kt;
    const __bf16* gb = gbCol + (size_t)(kt + br) * N;
    unsigned ao = buf ? aoff1 : aoff0;
    unsigned bo = buf ? boff1 : boff0;
    async_g2l_b128(ao,      ga);
    async_g2l_b128(ao + 16, ga + 8);
    async_g2l_b128(bo,      gb);
    async_g2l_b128(bo + 16, gb + 8);
  };

  issue(0, 0);
  int cur = 0;
  for (int kt = 0; kt < K; kt += 32) {
    const bool more = (kt + 32) < K;
    if (more) { issue(cur ^ 1, kt + 32); wait_async4(); }
    else      { wait_async0(); }
    __syncthreads();

    // A fragments (ISA 16-bit A 16x32 layout)
    v16bf af[2];
#pragma unroll
    for (int i = 0; i < 2; i++) {
      int m = wm * 32 + i * 16 + l16;
#pragma unroll
      for (int r = 0; r < 8; r++) {
        int k = ((r >> 2) << 4) + (half << 3) + ((r & 3) << 1);
        unsigned pk = *(const unsigned*)&As[cur][m][k];
        af[i][2 * r] = bflo(pk); af[i][2 * r + 1] = bfhi(pk);
      }
    }
#pragma unroll
    for (int j = 0; j < 4; j++) {
      v16bf bfr;
      int n = wn * 64 + j * 16 + l16;
#pragma unroll
      for (int r = 0; r < 8; r++) {
        int k0 = ((r >> 2) << 4) + (half << 3) + ((r & 3) << 1);
        bfr[2 * r]     = Bs[cur][k0][n];
        bfr[2 * r + 1] = Bs[cur][k0 + 1][n];
      }
#pragma unroll
      for (int i = 0; i < 2; i++)
        acc[i][j] = __builtin_amdgcn_wmma_f32_16x16x32_bf16(
            false, af[i], false, bfr, (short)0, acc[i][j], false, false);
    }
    __syncthreads();
    cur ^= 1;
  }

  // epilogue (C layout: VGPR r -> M = r + 8*half, N = lane&15)
#pragma unroll
  for (int i = 0; i < 2; i++)
#pragma unroll
    for (int j = 0; j < 4; j++)
#pragma unroll
      for (int r = 0; r < 8; r++) {
        int m = blockM + wm * 32 + i * 16 + r + (half << 3);
        int n = blockN + wn * 64 + j * 16 + l16;
        float v = acc[i][j][r] + bias[n];
        if (EPI == 0) {
          OutB[(size_t)m * N + n] = f2bf(v);
        } else if (EPI == 1) {
          float gl = 0.5f * v * (1.0f + erff(v * 0.70710678118654752f));
          OutB[(size_t)m * N + n] = f2bf(gl);
        } else if (EPI == 2) {
          float* p = Xres + (size_t)m * N + n;
          *p = *p + v;
        } else { // qkv scatter
          int b = m / TT, t = m - b * TT;
          int part = n / DD;
          int col = n - part * DD;
          int nh = col & (NHH - 1);
          int hd = col >> 4;                         // col = hd*NH + nh, NH=16
          __bf16* dst = (part == 0) ? qb : (part == 1) ? kbuf : vbuf;
          dst[(((size_t)(b * NHH + nh)) * TT + t) * HDD + hd] = f2bf(v);
        }
      }
}

// =====================================================================
// Flash attention, WMMA for QK^T and P*V, online softmax in fp32.
// One wave handles a 16-row query tile of one (b,head). Block = 8 waves.
// =====================================================================
__global__ __launch_bounds__(256)
void attention_kernel(const __bf16* __restrict__ qb, const __bf16* __restrict__ kbuf,
                      const __bf16* __restrict__ vbuf, __bf16* __restrict__ attb) {
  __shared__ __bf16 Pl[8][16][32];    // per-wave P tile (C->A relayout)

  const int tid = threadIdx.x, lane = tid & 31, wid = tid >> 5;
  const int bh = blockIdx.x;                // b*NH + nh
  const int b = bh >> 4, nh = bh & 15;
  const int q0 = blockIdx.y * 128 + wid * 16;
  const int half = lane >> 4, l16 = lane & 15;

  const __bf16* Qh = qb   + (size_t)bh * TT * HDD;
  const __bf16* Kh = kbuf + (size_t)bh * TT * HDD;
  const __bf16* Vh = vbuf + (size_t)bh * TT * HDD;

  // Q fragments: two k-steps over HD=64
  v16bf qa[2];
#pragma unroll
  for (int ks = 0; ks < 2; ks++)
#pragma unroll
    for (int r = 0; r < 8; r++) {
      int k = ks * 32 + ((r >> 2) << 4) + (half << 3) + ((r & 3) << 1);
      unsigned pk = *(const unsigned*)(Qh + (size_t)(q0 + l16) * HDD + k);
      qa[ks][2 * r] = bflo(pk); qa[ks][2 * r + 1] = bfhi(pk);
    }

  v8f atta[4];
#pragma unroll
  for (int j = 0; j < 4; j++)
#pragma unroll
    for (int e = 0; e < 8; e++) atta[j][e] = 0.f;
  float mrow[8], lrow[8];
#pragma unroll
  for (int r = 0; r < 8; r++) { mrow[r] = -INFINITY; lrow[r] = 0.f; }

  const int nkb = (q0 + 15) / 32 + 1;   // causal: only key blocks <= last row
  for (int kb2 = 0; kb2 < nkb; kb2++) {
    const int s0 = kb2 * 32;

    // S = Q K^T (16x32) : 2 n-tiles x 2 k-steps = 4 WMMA
    v8f sacc[2];
#pragma unroll
    for (int j = 0; j < 2; j++)
#pragma unroll
      for (int e = 0; e < 8; e++) sacc[j][e] = 0.f;
#pragma unroll
    for (int j = 0; j < 2; j++)
#pragma unroll
      for (int ks = 0; ks < 2; ks++) {
        v16bf bk;
#pragma unroll
        for (int r = 0; r < 8; r++) {
          int d = ks * 32 + ((r >> 2) << 4) + (half << 3) + ((r & 3) << 1);
          unsigned pk = *(const unsigned*)(Kh + (size_t)(s0 + j * 16 + l16) * HDD + d);
          bk[2 * r] = bflo(pk); bk[2 * r + 1] = bfhi(pk);
        }
        sacc[j] = __builtin_amdgcn_wmma_f32_16x16x32_bf16(
            false, qa[ks], false, bk, (short)0, sacc[j], false, false);
      }

    // online softmax per row (row lives in one VGPR over one 16-lane half)
    float corr[8];
#pragma unroll
    for (int r = 0; r < 8; r++) {
      int row = q0 + r + (half << 3);
      float s0v = sacc[0][r] * SCALE;
      float s1v = sacc[1][r] * SCALE;
      if (s0 + l16 > row)      s0v = -INFINITY;
      if (s0 + 16 + l16 > row) s1v = -INFINITY;
      float mx = fmaxf(s0v, s1v);
#pragma unroll
      for (int mk = 1; mk < 16; mk <<= 1) mx = fmaxf(mx, __shfl_xor(mx, mk, 16));
      float mnew = fmaxf(mrow[r], mx);
      float p0 = expf(s0v - mnew);
      float p1 = expf(s1v - mnew);
      float ps = p0 + p1;
#pragma unroll
      for (int mk = 1; mk < 16; mk <<= 1) ps += __shfl_xor(ps, mk, 16);
      float c = expf(mrow[r] - mnew);
      lrow[r] = lrow[r] * c + ps;
      mrow[r] = mnew;
      corr[r] = c;
      Pl[wid][r + (half << 3)][l16]      = f2bf(p0);
      Pl[wid][r + (half << 3)][16 + l16] = f2bf(p1);
    }
#pragma unroll
    for (int j = 0; j < 4; j++)
#pragma unroll
      for (int r = 0; r < 8; r++) atta[j][r] = atta[j][r] * corr[r];

    // wave-local LDS ordering (LDS is in-order per wave; this also fences codegen)
    asm volatile("s_wait_dscnt 0" ::: "memory");

    // P fragment (A layout) from LDS
    v16bf pa;
#pragma unroll
    for (int r = 0; r < 8; r++) {
      int k = ((r >> 2) << 4) + (half << 3) + ((r & 3) << 1);
      unsigned pk = *(const unsigned*)&Pl[wid][l16][k];
      pa[2 * r] = bflo(pk); pa[2 * r + 1] = bfhi(pk);
    }
    // att += P * V : 4 n-tiles over HD=64
#pragma unroll
    for (int j = 0; j < 4; j++) {
      v16bf bv;
#pragma unroll
      for (int r = 0; r < 8; r++) {
        int k0 = ((r >> 2) << 4) + (half << 3) + ((r & 3) << 1);
        bv[2 * r]     = Vh[(size_t)(s0 + k0) * HDD + j * 16 + l16];
        bv[2 * r + 1] = Vh[(size_t)(s0 + k0 + 1) * HDD + j * 16 + l16];
      }
      atta[j] = __builtin_amdgcn_wmma_f32_16x16x32_bf16(
          false, pa, false, bv, (short)0, atta[j], false, false);
    }
  }

  // write out: reference merge d = nh*HD + hd
#pragma unroll
  for (int j = 0; j < 4; j++)
#pragma unroll
    for (int r = 0; r < 8; r++) {
      int row = q0 + r + (half << 3);
      float v = atta[j][r] / lrow[r];
      attb[(size_t)(b * TT + row) * DD + nh * HDD + j * 16 + l16] = f2bf(v);
    }
}

// =====================================================================
// LM head: out[b, v] = dot(lnf_x[b,:], wte[v,:])  (bandwidth-bound, fp32)
// One wave per vocab row, x cached in LDS.
// =====================================================================
__global__ __launch_bounds__(256)
void lmhead_kernel(const __bf16* __restrict__ xf, const float* __restrict__ wte,
                   float* __restrict__ out) {
  __shared__ float sx[2 * DD];
  int tid = threadIdx.x;
  for (int c = tid; c < 2 * DD; c += 256) sx[c] = bf2f(xf[c]);
  __syncthreads();
  int lane = tid & 31, wid = tid >> 5;
  int v = blockIdx.x * 8 + wid;
  if (v >= VV) return;
  const float* wr = wte + (size_t)v * DD;
  float s0 = 0.f, s1 = 0.f;
  for (int d = lane; d < DD; d += 32) {
    float w = wr[d];
    s0 += w * sx[d];
    s1 += w * sx[DD + d];
  }
#pragma unroll
  for (int mk = 16; mk > 0; mk >>= 1) {
    s0 += __shfl_xor(s0, mk, 32);
    s1 += __shfl_xor(s1, mk, 32);
  }
  if (lane == 0) { out[v] = s0; out[VV + v] = s1; }
}

// =====================================================================
extern "C" void kernel_launch(void* const* d_in, const int* in_sizes, int n_in,
                              void* d_out, int out_size, void* d_ws, size_t ws_size,
                              hipStream_t stream) {
  (void)in_sizes; (void)n_in; (void)out_size; (void)ws_size;
  const int*   tokens = (const int*)d_in[0];
  const float* wte    = (const float*)d_in[1];
  const float* wpe    = (const float*)d_in[2];
  const float* ln1w   = (const float*)d_in[3];
  const float* ln1b   = (const float*)d_in[4];
  const float* attnw  = (const float*)d_in[5];
  const float* attnb  = (const float*)d_in[6];
  const float* aprojw = (const float*)d_in[7];
  const float* aprojb = (const float*)d_in[8];
  const float* ln2w   = (const float*)d_in[9];
  const float* ln2b   = (const float*)d_in[10];
  const float* fcw    = (const float*)d_in[11];
  const float* fcb    = (const float*)d_in[12];
  const float* projw  = (const float*)d_in[13];
  const float* projb  = (const float*)d_in[14];
  const float* lnfw   = (const float*)d_in[15];
  const float* lnfb   = (const float*)d_in[16];
  float* out = (float*)d_out;

  char* ws = (char*)d_ws;
  size_t off = 0;
  auto carve = [&](size_t bytes) -> char* {
    char* p = ws + off; off += (bytes + 255) & ~(size_t)255; return p;
  };
  float*  x     = (float*)carve((size_t)MM * DD * 4);              // residual stream
  __bf16* hbf   = (__bf16*)carve((size_t)MM * DD * 2);             // LN out (GEMM A)
  __bf16* qbuf  = (__bf16*)carve((size_t)BB * NHH * TT * HDD * 2); // [B,NH,T,HD]
  __bf16* kbuf  = (__bf16*)carve((size_t)BB * NHH * TT * HDD * 2);
  __bf16* vbuf  = (__bf16*)carve((size_t)BB * NHH * TT * HDD * 2);
  __bf16* attb  = (__bf16*)carve((size_t)MM * DD * 2);             // attention out
  __bf16* fcbf  = (__bf16*)carve((size_t)MM * FF * 2);             // gelu(fc) out
  __bf16* wbf   = (__bf16*)carve((size_t)DD * FF * 2);             // bf16 weight staging
  __bf16* lnfbf = (__bf16*)carve((size_t)BB * DD * 2);             // lnf of last tokens

  embed_kernel<<<MM, 256, 0, stream>>>(tokens, wte, wpe, x);

  auto cvt = [&](const float* src, int n) {
    cvt_f32_bf16_kernel<<<(n / 8 + 255) / 256, 256, 0, stream>>>(src, wbf, n);
  };

  for (int l = 0; l < LL; l++) {
    layernorm_kernel<<<MM, 256, 0, stream>>>(x, ln1w + l * DD, ln1b + l * DD, hbf, 0, 1);
    cvt(attnw + (size_t)l * DD * 3 * DD, DD * 3 * DD);
    gemm_bf16_kernel<3><<<dim3(3 * DD / 128, MM / 128), 256, 0, stream>>>(
        hbf, wbf, attnb + (size_t)l * 3 * DD,
        nullptr, nullptr, qbuf, kbuf, vbuf, MM, 3 * DD, DD);
    attention_kernel<<<dim3(BB * NHH, TT / 128), 256, 0, stream>>>(qbuf, kbuf, vbuf, attb);
    cvt(aprojw + (size_t)l * DD * DD, DD * DD);
    gemm_bf16_kernel<2><<<dim3(DD / 128, MM / 128), 256, 0, stream>>>(
        attb, wbf, aprojb + (size_t)l * DD,
        x, nullptr, nullptr, nullptr, nullptr, MM, DD, DD);
    layernorm_kernel<<<MM, 256, 0, stream>>>(x, ln2w + l * DD, ln2b + l * DD, hbf, 0, 1);
    cvt(fcw + (size_t)l * DD * FF, DD * FF);
    gemm_bf16_kernel<1><<<dim3(FF / 128, MM / 128), 256, 0, stream>>>(
        hbf, wbf, fcb + (size_t)l * FF,
        nullptr, fcbf, nullptr, nullptr, nullptr, MM, FF, DD);
    cvt(projw + (size_t)l * FF * DD, FF * DD);
    gemm_bf16_kernel<2><<<dim3(DD / 128, MM / 128), 256, 0, stream>>>(
        fcbf, wbf, projb + (size_t)l * DD,
        x, nullptr, nullptr, nullptr, nullptr, MM, DD, FF);
  }

  // final LN on only the two rows the output needs (t = T-1 for b = 0,1)
  layernorm_kernel<<<BB, 256, 0, stream>>>(x, lnfw, lnfb, lnfbf, TT - 1, TT);
  lmhead_kernel<<<(VV + 7) / 8, 256, 0, stream>>>(lnfbf, wte, out);
}
